// LSTM_48902497632713
// MI455X (gfx1250) — compile-verified, matrix operand order checked by reading, not run on hardware
//
#include <hip/hip_runtime.h>

typedef __attribute__((ext_vector_type(16))) __bf16 v16bf;
typedef __attribute__((ext_vector_type(8)))  float  v8f;

#define B_DIM   8192
#define IN_DIM  2048
#define OUT_DIM 2048
#define TILE_M  256            // batch rows per block
#define TILE_O  64             // output cols per block
#define EFFN    256            // 4 gates * TILE_O, column order n = o*4 + g
#define TILE_K  32
#define TOTAL_K (IN_DIM + OUT_DIM)
#define PITCH   48             // bf16 elems per LDS row (32 data + 16 pad, 96B)

union Pack8 { __bf16 h[8]; uint4 u; };

// 16x32 bf16 fragment from LDS stored [row][k] with row pitch PITCH.
// CDNA5 16-bit A layout: lanes 0-15 row M=lane, K={0..7,16..23};
// lanes 16-31 row M=lane-16, K={8..15,24..31}. B tiles are stored
// K-transposed ([n][k]) so the same loader serves both operands.
__device__ __forceinline__ v16bf load_frag(const __bf16* lds, int rowBase, int lane) {
  const int r  = rowBase + (lane & 15);
  const int kb = (lane >> 4) * 8;          // 0 or 8 (bf16 elems)
  Pack8 a, b;
  a.u = *reinterpret_cast<const uint4*>(lds + r * PITCH + kb);
  b.u = *reinterpret_cast<const uint4*>(lds + r * PITCH + kb + 16);
  v16bf out;
#pragma unroll
  for (int i = 0; i < 8; ++i) { out[i] = a.h[i]; out[i + 8] = b.h[i]; }
  return out;
}

__device__ __forceinline__ float sigf(float v) {
  return 1.0f / (1.0f + __expf(-v));
}

__global__ __launch_bounds__(512) void lstm_fused(
    const float* __restrict__ x,      // [B, IN]
    const float* __restrict__ prev_h, // [B, OUT]
    const float* __restrict__ prev_c, // [B, OUT]
    const float* __restrict__ Wi,     // [4, IN, OUT]
    const float* __restrict__ Wh,     // [4, OUT, OUT]
    const float* __restrict__ bi,     // [4, OUT]
    const float* __restrict__ bh,     // [4, OUT]
    float* __restrict__ next_h,       // [B, OUT]
    float* __restrict__ next_c)       // [B, OUT]
{
  __shared__ __bf16 ldsA[TILE_M * PITCH];  // 24 KB, [row][k]
  __shared__ __bf16 ldsB[EFFN  * PITCH];   // 24 KB, [n][k], n = o*4+g

  const int tid  = threadIdx.x;
  const int lane = tid & 31;
  const int wave = tid >> 5;               // 0..15

  const int m0 = blockIdx.y * TILE_M;
  const int o0 = blockIdx.x * TILE_O;

  // 16 waves tile the 256x256 block as 8 (M) x 2 (N): each wave 32x128.
  const int mrow  = (wave >> 1) * 32;
  const int nbase = (wave & 1) * 128;

  v8f acc[2][8];
#pragma unroll
  for (int mi = 0; mi < 2; ++mi)
#pragma unroll
    for (int ni = 0; ni < 8; ++ni)
      acc[mi][ni] = (v8f)(0.0f);

  // --- staging assignments (fixed per thread) ---
  const int aRow = tid >> 1;               // 0..255
  const int aKh  = (tid & 1) * 16;         // 0 or 16
  const int bN   = tid & 255;              // eff column n = o*4 + g
  const int bKh  = (tid >> 8) * 16;        // 0 or 16
  const int bG   = bN & 3;                 // gate
  const int bO   = bN >> 2;                // col within gate (0..63)

  const size_t gateStride = (size_t)IN_DIM * OUT_DIM;

  // Register double-buffer for the software pipeline.
  float4 fa[4];
  float  fb[16];

  // Issue the global loads for chunk starting at kknext into fa/fb.
  auto issue_loads = [&](int kknext) {
    const bool ph = kknext >= IN_DIM;
    const int  k0 = ph ? (kknext - IN_DIM) : kknext;
    const float* __restrict__ Asrc = ph ? prev_h : x;
    const float* __restrict__ Wsrc = (ph ? Wh : Wi) + (size_t)bG * gateStride;

    const float* sa = Asrc + (size_t)(m0 + aRow) * IN_DIM + k0 + aKh;
    __builtin_prefetch(sa + TILE_K, 0, 0);   // 2 chunks ahead into L2
    fa[0] = *reinterpret_cast<const float4*>(sa + 0);
    fa[1] = *reinterpret_cast<const float4*>(sa + 4);
    fa[2] = *reinterpret_cast<const float4*>(sa + 8);
    fa[3] = *reinterpret_cast<const float4*>(sa + 12);

    const float* sb = Wsrc + (size_t)(k0 + bKh) * OUT_DIM + o0 + bO;
    __builtin_prefetch(sb + (size_t)TILE_K * OUT_DIM, 0, 0);
#pragma unroll
    for (int i = 0; i < 16; ++i) fb[i] = sb[(size_t)i * OUT_DIM];
  };

  issue_loads(0);  // prime the pipeline

  for (int kk = 0; kk < TOTAL_K; kk += TILE_K) {
    // ---- pack current chunk's registers -> LDS ----
    {
      Pack8 p0, p1;
      const float* f = reinterpret_cast<const float*>(fa);
#pragma unroll
      for (int i = 0; i < 8; ++i) { p0.h[i] = (__bf16)f[i]; p1.h[i] = (__bf16)f[i + 8]; }
      __bf16* d = &ldsA[aRow * PITCH + aKh];
      *reinterpret_cast<uint4*>(d)     = p0.u;
      *reinterpret_cast<uint4*>(d + 8) = p1.u;
    }
    {
      Pack8 p0, p1;
#pragma unroll
      for (int i = 0; i < 8; ++i) { p0.h[i] = (__bf16)fb[i]; p1.h[i] = (__bf16)fb[i + 8]; }
      __bf16* d = &ldsB[bN * PITCH + bKh];
      *reinterpret_cast<uint4*>(d)     = p0.u;
      *reinterpret_cast<uint4*>(d + 8) = p1.u;
    }
    __syncthreads();

    // ---- issue next chunk's global loads (latency hides under compute) ----
    if (kk + TILE_K < TOTAL_K) issue_loads(kk + TILE_K);

    // ---- compute: 16 WMMA per wave per chunk, B-frag rotated one ahead ----
    v16bf af0 = load_frag(ldsA, mrow,      lane);
    v16bf af1 = load_frag(ldsA, mrow + 16, lane);
    v16bf bf_cur = load_frag(ldsB, nbase, lane);
#pragma unroll
    for (int ni = 0; ni < 8; ++ni) {
      v16bf bf_nxt = (ni < 7) ? load_frag(ldsB, nbase + (ni + 1) * 16, lane)
                              : bf_cur;
      acc[0][ni] = __builtin_amdgcn_wmma_f32_16x16x32_bf16(
          false, af0, false, bf_cur, (short)0, acc[0][ni], false, false);
      acc[1][ni] = __builtin_amdgcn_wmma_f32_16x16x32_bf16(
          false, af1, false, bf_cur, (short)0, acc[1][ni], false, false);
      bf_cur = bf_nxt;
    }
    __syncthreads();
  }

  // ---- fused LSTM epilogue ----
  // C/D layout: lanes 0-15 -> col = tile base + lane, row = base + v;
  // lanes 16-31 -> row += 8. Column order n = o*4 + g means each lane quad
  // holds the 4 gates of one (row, o): combine with intra-quad shuffles.
  const int nl    = lane & 15;
  const int radd  = (lane >> 4) * 8;
  const int qbase = lane & ~3;
#pragma unroll
  for (int ni = 0; ni < 8; ++ni) {
    const int nloc = nbase + ni * 16 + nl;   // eff col
    const int g    = nloc & 3;
    const int oc   = o0 + (nloc >> 2);       // global output col
    const float bsum = bi[g * OUT_DIM + oc] + bh[g * OUT_DIM + oc];
#pragma unroll
    for (int mi = 0; mi < 2; ++mi) {
#pragma unroll
      for (int v = 0; v < 8; ++v) {
        const int row = m0 + mrow + mi * 16 + radd + v;
        const float val = acc[mi][ni][v] + bsum;
        const float gi  = __shfl(val, qbase + 0, 32);  // in gate
        const float gf  = __shfl(val, qbase + 1, 32);  // forget gate
        const float gi2 = __shfl(val, qbase + 2, 32);  // in gate 2
        const float go  = __shfl(val, qbase + 3, 32);  // out gate
        if (g < 2) {
          const size_t idx = (size_t)row * OUT_DIM + oc;
          const float c = sigf(gf) * prev_c[idx] + sigf(gi) * sigf(gi2);
          if (g == 0) next_h[idx] = sigf(go) + tanhf(c);   // faithful: add
          else        next_c[idx] = c;
        }
      }
    }
  }
}

extern "C" void kernel_launch(void* const* d_in, const int* in_sizes, int n_in,
                              void* d_out, int out_size, void* d_ws, size_t ws_size,
                              hipStream_t stream) {
  (void)in_sizes; (void)n_in; (void)out_size; (void)d_ws; (void)ws_size;
  const float* x      = (const float*)d_in[0];
  const float* prev_h = (const float*)d_in[1];
  const float* prev_c = (const float*)d_in[2];
  const float* Wi     = (const float*)d_in[3];
  const float* Wh     = (const float*)d_in[4];
  const float* bi     = (const float*)d_in[5];
  const float* bh     = (const float*)d_in[6];
  float* next_h = (float*)d_out;
  float* next_c = next_h + (size_t)B_DIM * OUT_DIM;

  dim3 grid(OUT_DIM / TILE_O, B_DIM / TILE_M);  // 32 x 32 blocks
  lstm_fused<<<grid, 512, 0, stream>>>(x, prev_h, prev_c, Wi, Wh, bi, bh,
                                       next_h, next_c);
}
